// CausalMultiHeadSelfAttention_71064528879764
// MI455X (gfx1250) — matrix operand order repaired and checked by reading, not compiled, not run on hardware
//
#include <hip/hip_runtime.h>
#include <hip/hip_bf16.h>

// ---------------------------------------------------------------------------
// Types for CDNA5 WMMA (wave32): bf16 A/B (16 halves = 8 VGPRs), f32 C/D (8 VGPRs)
// ---------------------------------------------------------------------------
typedef __attribute__((ext_vector_type(16))) __bf16 bf16x16;
typedef __attribute__((ext_vector_type(8)))  float  f32x8;
typedef __attribute__((ext_vector_type(4)))  unsigned int u32x4;
typedef __attribute__((ext_vector_type(8)))  int    i32x8;
typedef __attribute__((ext_vector_type(4)))  int    i32x4;

union FragBF { unsigned short u[16]; bf16x16 v; };

__device__ __forceinline__ unsigned short f32_to_bf16(float f) {
    union { float f; unsigned int u; } c; c.f = f;
    unsigned int r = c.u + 0x7FFFu + ((c.u >> 16) & 1u);   // round-to-nearest-even
    return (unsigned short)(r >> 16);
}

__device__ __forceinline__ f32x8 zero8() {
    f32x8 z = {0.f,0.f,0.f,0.f,0.f,0.f,0.f,0.f};
    return z;
}

// A-matrix fragment (16x32 bf16). Lane L holds row M=L&15; this code's contiguous K order (same as B):
// VGPR v, half h -> K = 2v + h + (lane>=16 ? 16 : 0)
__device__ __forceinline__ bf16x16 load_frag_a(const unsigned short* rowp, int laneHi) {
    FragBF f;
#pragma unroll
    for (int v = 0; v < 8; ++v) {
        int k = 2 * v + (laneHi ? 16 : 0);
        f.u[2 * v]     = rowp[k];
        f.u[2 * v + 1] = rowp[k + 1];
    }
    return f.v;
}

// B-matrix fragment (32x16 bf16): lane holds N=lane&15, 16 contiguous K halves.
__device__ __forceinline__ bf16x16 load_frag_b(const unsigned short* p) {
    FragBF f;
#pragma unroll
    for (int j = 0; j < 16; ++j) f.u[j] = p[j];
    return f.v;
}

__device__ __forceinline__ f32x8 wmma_bf16(bf16x16 a, bf16x16 b, f32x8 c) {
    return __builtin_amdgcn_wmma_f32_16x16x32_bf16(false, a, false, b,
                                                   (short)0, c, false, false);
}

// ---------------------------------------------------------------------------
// Tensor Data Mover: 2D tile load Global->LDS with hardware LDS row padding.
// D# per CDNA5 ISA:
//   group0: count=1 | lds_addr | global_addr[56:0] | type=2
//   group1: data_size=2B, pad_enable, pad_interval=4 (32 DW = one 64-half row),
//           pad_amount=3 (4 DW = 8 halves) -> LDS row stride 72 halves.
// Issued by one wave; completion via s_wait_tensorcnt.
// This toolchain exposes the 6-arg builtin:
//   (u32x4 g0, i32x8 g1, i32x4 g2, i32x4 g3, i32x8, i32 cpol)
// ---------------------------------------------------------------------------
__device__ __forceinline__ void tdm_load_2d_bf16(
    unsigned lds_off, const unsigned short* gptr,
    unsigned tile_d0, unsigned tile_d1,
    unsigned long long row_stride_elems, unsigned tensor_d1)
{
    unsigned long long ga = (unsigned long long)(uintptr_t)gptr;
    u32x4 g0;
    g0[0] = 1u;                                            // count=1 (valid user D#)
    g0[1] = lds_off;                                       // lds_addr (bytes)
    g0[2] = (unsigned)(ga & 0xFFFFFFFFu);                  // global_addr[31:0]
    g0[3] = (unsigned)((ga >> 32) & 0x01FFFFFFu) | (2u << 30); // addr[56:32] | type=2

    const unsigned w0 = (1u << 16)   // data_size = 2 bytes
                      | (1u << 20)   // pad_enable
                      | (4u << 22)   // pad_interval: 32 DWORDs (one 128B row)
                      | (3u << 25);  // pad_amount:  4 DWORDs (16B = 8 halves)
    unsigned long long s0 = row_stride_elems;
    i32x8 g1;
    g1[0] = (int)w0;
    g1[1] = (int)((tile_d0 & 0xFFFFu) << 16);              // tensor_dim0 lo16 (<<16), atomic_addr=0
    g1[2] = (int)((tile_d0 >> 16) | ((tensor_d1 & 0xFFFFu) << 16)); // dim0 hi | dim1 lo
    g1[3] = (int)((tensor_d1 >> 16) | (tile_d0 << 16));    // dim1 hi | tile_dim0
    g1[4] = (int)(tile_d1 & 0xFFFFu);                      // tile_dim1 (tile_dim2=0)
    g1[5] = (int)(s0 & 0xFFFFFFFFu);                       // tensor_dim0_stride lo32
    g1[6] = (int)((s0 >> 32) & 0xFFFFu);                   // stride hi16 (dim1_stride=0)
    g1[7] = 0;
    i32x4 z4 = {0, 0, 0, 0};
    i32x8 z8 = {0, 0, 0, 0, 0, 0, 0, 0};
    __builtin_amdgcn_tensor_load_to_lds(g0, g1, z4, z4, z8, 0);
}

// ---------------------------------------------------------------------------
// GEMM: Y[m][n] = sum_k X[m][k] * W[n][k]   (M=4096, N=1024, K=1024, fp32 I/O,
// bf16 WMMA with f32 accumulate). 256 threads = 8 waves; block tile 128x128;
// wave tile 32x64 (2x4 accumulators); K step 32; double-buffered LDS so the
// next tile's global_load_b128 + convert co-executes with the WMMAs.
// ---------------------------------------------------------------------------
__global__ __launch_bounds__(256) void gemm_xwT_bf16(
    const float* __restrict__ X, const float* __restrict__ W, float* __restrict__ Y)
{
    constexpr int LD = 40; // 32 K-halves + 8 pad (80B row stride -> 20 banks)
    __shared__ alignas(16) unsigned short At[2][128 * LD];
    __shared__ alignas(16) unsigned short Bt[2][128 * LD];

    const int tid    = threadIdx.x;
    const int lane   = tid & 31;
    const int laneHi = lane >> 4;
    const int l15    = lane & 15;
    const int wid    = tid >> 5;
    const int wm     = wid & 3;   // 4 waves along M
    const int wn     = wid >> 2;  // 2 waves along N
    const int m0     = blockIdx.y * 128;
    const int n0     = blockIdx.x * 128;

    f32x8 acc[2][4];
#pragma unroll
    for (int i = 0; i < 2; ++i)
#pragma unroll
        for (int j = 0; j < 4; ++j) acc[i][j] = zero8();

    const int lr = tid >> 1;        // 0..127 : tile row loaded by this thread
    const int lk = (tid & 1) * 16;  // 0 or 16 : K half

    auto load_tile = [&](int kt, int buf) {
        const float4* xs = (const float4*)(X + (size_t)(m0 + lr) * 1024 + kt + lk);
        const float4* ws = (const float4*)(W + (size_t)(n0 + lr) * 1024 + kt + lk);
#pragma unroll
        for (int q = 0; q < 4; ++q) {
            float4 xa = xs[q];
            float4 wa = ws[q];
            int kk = lk + q * 4;
            At[buf][lr * LD + kk + 0] = f32_to_bf16(xa.x);
            At[buf][lr * LD + kk + 1] = f32_to_bf16(xa.y);
            At[buf][lr * LD + kk + 2] = f32_to_bf16(xa.z);
            At[buf][lr * LD + kk + 3] = f32_to_bf16(xa.w);
            Bt[buf][lr * LD + kk + 0] = f32_to_bf16(wa.x);
            Bt[buf][lr * LD + kk + 1] = f32_to_bf16(wa.y);
            Bt[buf][lr * LD + kk + 2] = f32_to_bf16(wa.z);
            Bt[buf][lr * LD + kk + 3] = f32_to_bf16(wa.w);
        }
    };

    load_tile(0, 0);
    for (int it = 0; it < 32; ++it) {
        __syncthreads();   // tile(it) in LDS; buf (it+1)&1 free for reuse
        if (it + 1 < 32) load_tile((it + 1) * 32, (it + 1) & 1);

        const unsigned short* A = At[it & 1];
        const unsigned short* Bb = Bt[it & 1];
        bf16x16 afrag[2];
        afrag[0] = load_frag_a(A + (wm * 32 + 0  + l15) * LD, laneHi);
        afrag[1] = load_frag_a(A + (wm * 32 + 16 + l15) * LD, laneHi);
#pragma unroll
        for (int fn = 0; fn < 4; ++fn) {
            bf16x16 bfrag = load_frag_b(Bb + (wn * 64 + fn * 16 + l15) * LD + laneHi * 16);
            acc[0][fn] = wmma_bf16(afrag[0], bfrag, acc[0][fn]);
            acc[1][fn] = wmma_bf16(afrag[1], bfrag, acc[1][fn]);
        }
    }

    // Epilogue: C/D layout -> VGPR r holds (row r or r+8, col lane&15)
#pragma unroll
    for (int fr = 0; fr < 2; ++fr)
#pragma unroll
        for (int fn = 0; fn < 4; ++fn)
#pragma unroll
            for (int r = 0; r < 8; ++r) {
                int row = m0 + wm * 32 + fr * 16 + r + 8 * laneHi;
                int col = n0 + wn * 64 + fn * 16 + l15;
                Y[(size_t)row * 1024 + col] = acc[fr][fn][r];
            }
}

// ---------------------------------------------------------------------------
// RoPE + bf16 convert + head-major relayout: P[b,s,h*64+2i..] -> Out[b,h,s,2i..]
// ---------------------------------------------------------------------------
__global__ __launch_bounds__(256) void rope_to_bf16(
    const float* __restrict__ P, unsigned short* __restrict__ Out,
    const int* __restrict__ tpos, const int* __restrict__ theta_p)
{
    int idx = blockIdx.x * blockDim.x + threadIdx.x; // 2^21 threads
    if (idx >= (1 << 21)) return;
    int i = idx & 31;
    int s = (idx >> 5) & 2047;
    int h = (idx >> 16) & 15;
    int b = idx >> 20;

    float theta = (float)theta_p[0];
    float pos   = (float)tpos[s];
    float fr    = __expf(-((float)(2 * i) / 64.f) * __logf(theta));
    float ang   = pos * fr;
    float sn, cs;
    __sincosf(ang, &sn, &cs);

    size_t src = ((size_t)(b * 2048 + s)) * 1024 + h * 64 + 2 * i;
    float e = P[src], o = P[src + 1];
    size_t dst = (((size_t)(b * 16 + h)) * 2048 + s) * 64 + 2 * i;
    Out[dst]     = f32_to_bf16(e * cs - o * sn);
    Out[dst + 1] = f32_to_bf16(e * sn + o * cs);
}

// V: [b,s,h*64+d] fp32 -> Vt[b,h,d,s] bf16 (key-contiguous for PV B-fragments)
__global__ __launch_bounds__(256) void v_to_bf16_t(
    const float* __restrict__ V, unsigned short* __restrict__ Vt)
{
    int idx = blockIdx.x * blockDim.x + threadIdx.x; // 2^22 threads
    if (idx >= (1 << 22)) return;
    int s = idx & 2047;
    int d = (idx >> 11) & 63;
    int h = (idx >> 17) & 15;
    int b = idx >> 21;
    Vt[idx] = f32_to_bf16(V[((size_t)(b * 2048 + s)) * 1024 + h * 64 + d]);
}

// ---------------------------------------------------------------------------
// Flash attention (causal, online softmax, bf16 WMMA, fp32 accumulate).
// Block = 128 threads (4 waves); 64 queries/block (16 per wave); 64-key blocks.
// K/V tiles staged into LDS by the Tensor Data Mover (wave 0 issues, TDM pads
// LDS rows to 72 halves in hardware); completion via s_wait_tensorcnt + barrier.
// Qh,Kh: [B,H,S,64] bf16.  Vt: [B,H,64,S] bf16.  vals: [B,S,1024] fp32.
// ---------------------------------------------------------------------------
__global__ __launch_bounds__(128) void flash_attn(
    const unsigned short* __restrict__ Qh, const unsigned short* __restrict__ Kh,
    const unsigned short* __restrict__ Vt, float* __restrict__ vals)
{
    constexpr int STR = 72; // 64 halves + 8 pad (144B rows, bank-conflict free)
    __shared__ alignas(16) unsigned short Qs[64 * STR];
    __shared__ alignas(16) unsigned short Ks[64 * STR];
    __shared__ alignas(16) unsigned short Vs[64 * STR];      // [d][key]
    __shared__ alignas(16) unsigned short Ps[4][16 * STR];   // per-wave P tile

    const int tid    = threadIdx.x;
    const int lane   = tid & 31;
    const int laneHi = lane >> 4;
    const int l15    = lane & 15;
    const int w      = tid >> 5;

    const int bh = blockIdx.y;          // b*16 + h
    const int qt = blockIdx.x;          // query tile (64 rows)
    const int q0 = qt * 64;
    const size_t baseQK = (size_t)bh * 2048 * 64;
    const size_t baseV  = (size_t)bh * 64 * 2048;

    // Stage Q tile (64x64 bf16) into LDS, then build per-wave A-fragments
    {
        int r = tid >> 1;
        int hoff = (tid & 1) * 32;
        const uint4* src = (const uint4*)(Qh + baseQK + (size_t)(q0 + r) * 64 + hoff);
        uint4* dst = (uint4*)(Qs + r * STR + hoff);
#pragma unroll
        for (int q = 0; q < 4; ++q) dst[q] = src[q];
    }
    __syncthreads();
    bf16x16 qa[2];
    qa[0] = load_frag_a(Qs + (w * 16 + l15) * STR + 0,  laneHi);
    qa[1] = load_frag_a(Qs + (w * 16 + l15) * STR + 32, laneHi);

    f32x8 o[4];
#pragma unroll
    for (int jd = 0; jd < 4; ++jd) o[jd] = zero8();
    float m_i[8], l_i[8];
#pragma unroll
    for (int r = 0; r < 8; ++r) { m_i[r] = -INFINITY; l_i[r] = 0.f; }

    const unsigned ldsK = (unsigned)(uintptr_t)(void*)Ks;  // LDS byte offset (low 32b)
    const unsigned ldsV = (unsigned)(uintptr_t)(void*)Vs;

    const int nj = qt + 1;  // causal: key blocks 0..qt
    for (int j = 0; j < nj; ++j) {
        const int kv0 = j * 64;
        __syncthreads();  // all waves done reading previous K/V tiles
        if (w == 0) {
            // K tile: 64 rows x 64 halves, row stride 64 (contiguous rows)
            tdm_load_2d_bf16(ldsK, Kh + baseQK + (size_t)kv0 * 64, 64, 64, 64, 64);
            // V tile: 64 d-rows x 64 keys, row stride 2048
            tdm_load_2d_bf16(ldsV, Vt + baseV + kv0, 64, 64, 2048, 64);
            __builtin_amdgcn_s_wait_tensorcnt(0);
        }
        __syncthreads();  // publish TDM-written LDS to all waves

        // scores S = Q K^T  (16 rows x 64 keys per wave; d_k=64 -> 2 K-steps)
        f32x8 s[4];
#pragma unroll
        for (int jn = 0; jn < 4; ++jn) s[jn] = zero8();
#pragma unroll
        for (int ks = 0; ks < 2; ++ks)
#pragma unroll
            for (int jn = 0; jn < 4; ++jn) {
                bf16x16 kb = load_frag_b(Ks + (jn * 16 + l15) * STR + ks * 32 + laneHi * 16);
                s[jn] = wmma_bf16(qa[ks], kb, s[jn]);
            }

        // scale + causal mask (only the diagonal block needs masking)
        const float sc = 0.125f; // 1/sqrt(64)
        if (j == qt) {
#pragma unroll
            for (int jn = 0; jn < 4; ++jn)
#pragma unroll
                for (int r = 0; r < 8; ++r) {
                    int key  = kv0 + jn * 16 + l15;
                    int qrow = q0 + w * 16 + r + 8 * laneHi;
                    s[jn][r] = (key <= qrow) ? s[jn][r] * sc : -INFINITY;
                }
        } else {
#pragma unroll
            for (int jn = 0; jn < 4; ++jn)
#pragma unroll
                for (int r = 0; r < 8; ++r) s[jn][r] *= sc;
        }

        // online softmax: row max across 4 accs + 16-lane shuffle reduction
        float alpha[8];
#pragma unroll
        for (int r = 0; r < 8; ++r) {
            float v = fmaxf(fmaxf(s[0][r], s[1][r]), fmaxf(s[2][r], s[3][r]));
            v = fmaxf(v, __shfl_xor(v, 1, 32));
            v = fmaxf(v, __shfl_xor(v, 2, 32));
            v = fmaxf(v, __shfl_xor(v, 4, 32));
            v = fmaxf(v, __shfl_xor(v, 8, 32));
            float mn = fmaxf(m_i[r], v);
            alpha[r] = __expf(m_i[r] - mn);
            m_i[r]   = mn;
        }

        // P = exp(S - m): store bf16 to LDS (C-layout -> matrix), rowsum
        float rs[8];
#pragma unroll
        for (int r = 0; r < 8; ++r) rs[r] = 0.f;
#pragma unroll
        for (int jn = 0; jn < 4; ++jn)
#pragma unroll
            for (int r = 0; r < 8; ++r) {
                float p = __expf(s[jn][r] - m_i[r]);
                rs[r] += p;
                Ps[w][(r + 8 * laneHi) * STR + jn * 16 + l15] = f32_to_bf16(p);
            }
#pragma unroll
        for (int r = 0; r < 8; ++r) {
            float v = rs[r];
            v += __shfl_xor(v, 1, 32);
            v += __shfl_xor(v, 2, 32);
            v += __shfl_xor(v, 4, 32);
            v += __shfl_xor(v, 8, 32);
            l_i[r] = l_i[r] * alpha[r] + v;
        }

        // rescale running O, then O += P V  (contraction over 64 keys)
#pragma unroll
        for (int jd = 0; jd < 4; ++jd)
#pragma unroll
            for (int r = 0; r < 8; ++r) o[jd][r] *= alpha[r];

#pragma unroll
        for (int ks = 0; ks < 2; ++ks) {
            bf16x16 pa = load_frag_a(Ps[w] + l15 * STR + ks * 32, laneHi);
#pragma unroll
            for (int jd = 0; jd < 4; ++jd) {
                bf16x16 vb = load_frag_b(Vs + (jd * 16 + l15) * STR + ks * 32 + laneHi * 16);
                o[jd] = wmma_bf16(pa, vb, o[jd]);
            }
        }
    }

    // epilogue: O / l -> vals[b, q, h*64 + d]
    const int b = bh >> 4, h = bh & 15;
#pragma unroll
    for (int jd = 0; jd < 4; ++jd)
#pragma unroll
        for (int r = 0; r < 8; ++r) {
            int qrow = q0 + w * 16 + r + 8 * laneHi;
            vals[((size_t)(b * 2048 + qrow)) * 1024 + h * 64 + jd * 16 + l15] =
                o[jd][r] * (1.f / l_i[r]);
        }
}

// ---------------------------------------------------------------------------
// Launcher
// ---------------------------------------------------------------------------
extern "C" void kernel_launch(void* const* d_in, const int* in_sizes, int n_in,
                              void* d_out, int out_size, void* d_ws, size_t ws_size,
                              hipStream_t stream)
{
    (void)in_sizes; (void)n_in; (void)out_size; (void)ws_size;
    const float* x    = (const float*)d_in[0];
    const float* Wq   = (const float*)d_in[1];
    const float* Wk   = (const float*)d_in[2];
    const float* Wv   = (const float*)d_in[3];
    const float* Wo   = (const float*)d_in[4];
    const int* tpos   = (const int*)d_in[5];
    const int* thetap = (const int*)d_in[6];

    const size_t MD = 4096ull * 1024ull;
    float* Qf   = (float*)d_ws;
    float* Kf   = Qf + MD;
    float* Vf   = Kf + MD;
    float* vals = Vf + MD;
    unsigned short* Qh  = (unsigned short*)(vals + MD);
    unsigned short* Kh  = Qh + MD;
    unsigned short* Vtb = Kh + MD;

    dim3 gg(8, 32), gb(256);
    gemm_xwT_bf16<<<gg, gb, 0, stream>>>(x, Wq, Qf);
    gemm_xwT_bf16<<<gg, gb, 0, stream>>>(x, Wk, Kf);
    gemm_xwT_bf16<<<gg, gb, 0, stream>>>(x, Wv, Vf);

    rope_to_bf16<<<(1 << 21) / 256, 256, 0, stream>>>(Qf, Qh, tpos, thetap);
    rope_to_bf16<<<(1 << 21) / 256, 256, 0, stream>>>(Kf, Kh, tpos, thetap);
    v_to_bf16_t<<<(1 << 22) / 256, 256, 0, stream>>>(Vf, Vtb);

    flash_attn<<<dim3(32, 32), 128, 0, stream>>>(Qh, Kh, Vtb, vals);

    gemm_xwT_bf16<<<gg, gb, 0, stream>>>(vals, Wo, (float*)d_out);
}
